// MultiHeadLatentAttention_3178275799417
// MI455X (gfx1250) — matrix operand order repaired and checked by reading, not compiled, not run on hardware
//
#include <hip/hip_runtime.h>
#include <hip/hip_bf16.h>

typedef __attribute__((ext_vector_type(16))) __bf16 v16bf;
typedef __attribute__((ext_vector_type(8)))  float  v8f;
typedef __attribute__((ext_vector_type(4)))  unsigned int u32x4;
typedef __attribute__((ext_vector_type(8)))  int  i32x8;
typedef __attribute__((ext_vector_type(4)))  int  i32x4;

constexpr int Bc = 2, Sc = 2048, Hc = 2048;
constexpr int NHc = 16, HDc = 128;
constexpr int NLHc = 4, LDc = 512;

// =====================================================================
// Generic WMMA GEMM:  Y[M,N] = X[M,K] @ W[N,K]^T + bias   (torch Linear)
// Block = 256 threads (8 waves), 128x128 output tile, K stepped by 32.
// =====================================================================
template<typename TX, typename TY>
__global__ __launch_bounds__(256)
void gemm_bias_wmma(const TX* __restrict__ X, const float* __restrict__ W,
                    const float* __restrict__ bias, TY* __restrict__ Y,
                    int M, int N, int K)
{
  constexpr int TM = 128, TN = 128, TK = 32;
  __shared__ __bf16 As[TM][TK + 2];
  __shared__ __bf16 Bs[TN][TK + 2];

  const int bn = blockIdx.x, bm = blockIdx.y;
  const int tid  = threadIdx.x;
  const int wave = tid >> 5;
  const int lane = tid & 31;
  const int half = lane >> 4;
  const int l16  = lane & 15;
  const int row0 = bm * TM, col0 = bn * TN;

  v8f acc[8] = {};

  for (int k0 = 0; k0 < K; k0 += TK) {
    __syncthreads();
    #pragma unroll
    for (int i = tid; i < TM * TK; i += 256) {
      const int r = i >> 5, c = i & 31;
      As[r][c] = (__bf16)(float)X[(size_t)(row0 + r) * K + (k0 + c)];
    }
    #pragma unroll
    for (int i = tid; i < TN * TK; i += 256) {
      const int r = i >> 5, c = i & 31;
      Bs[r][c] = (__bf16)W[(size_t)(col0 + r) * K + (k0 + c)];
    }
    if (k0 + TK < K) {   // gfx1250 global_prefetch for the next K tile
      __builtin_prefetch(&X[(size_t)(row0 + (tid >> 1)) * K + (k0 + TK)], 0, 1);
      __builtin_prefetch(&W[(size_t)(col0 + (tid >> 1)) * K + (k0 + TK)], 0, 1);
    }
    __syncthreads();

    // A fragment: 16x32, rows = this wave's 16-row strip
    v16bf afrag;
    #pragma unroll
    for (int j = 0; j < 16; ++j) {
      const int kk = ((j >> 3) << 4) + (half << 3) + (j & 7);
      afrag[j] = As[(wave << 4) + l16][kk];
    }
    #pragma unroll
    for (int t = 0; t < 8; ++t) {
      v16bf bfrag;   // B fragment: 32x16, B[k][n] = W[col0+n][k0+k]
      #pragma unroll
      for (int j = 0; j < 16; ++j)
        bfrag[j] = Bs[(t << 4) + l16][((j >> 3) << 4) + (half << 3) + (j & 7)];
      acc[t] = __builtin_amdgcn_wmma_f32_16x16x32_bf16(
          false, afrag, false, bfrag, (short)0, acc[t], false, false);
    }
  }

  // C layout: VGPR r -> row r (lanes 0-15) / row r+8 (lanes 16-31), N = lane%16
  #pragma unroll
  for (int t = 0; t < 8; ++t) {
    const int col = col0 + (t << 4) + l16;
    const float bv = bias[col];
    #pragma unroll
    for (int r = 0; r < 8; ++r) {
      const int row = row0 + (wave << 4) + r + (half << 3);
      Y[(size_t)row * N + col] = (TY)(acc[t][r] + bv);
    }
  }
}

// =====================================================================
// TDM: async DMA of one 32-row x 256-byte tile (rows strided by 1024 B)
// from global memory into LDS, with 1-dword padding after each 64-dword
// row (matches the [32][HDc+2] LDS layout).  D# per CDNA5 ISA Ch.8.
// =====================================================================
__device__ __forceinline__ void tdm_load_tile_32x128bf16(
    const __bf16* gsrc, unsigned lds_byte_off, int row_stride_dwords)
{
  const unsigned long long ga = (unsigned long long)(const void*)gsrc;

  u32x4 g0;
  g0[0] = 1u;                                   // count=1, user mode, no gather
  g0[1] = lds_byte_off;                         // lds_addr
  g0[2] = (unsigned)(ga & 0xffffffffu);         // global_addr[31:0]
  g0[3] = (unsigned)((ga >> 32) & 0x01ffffffu)  // global_addr[56:32]
        | (2u << 30);                           // type = 2 ("image")

  i32x8 g1;
  // data_size=2 (4B) | pad_enable | pad_interval=64 dwords (enc 5) | pad=1 dword
  g1[0] = (2 << 16) | (1 << 20) | (5 << 22) | (0 << 25);
  g1[1] = (64 << 16);          // tensor_dim0[15:0]=64 dwords  (bits 63:48)
  g1[2] = (32 << 16);          // tensor_dim0 hi=0; tensor_dim1 lo=32 (bits 95:80)
  g1[3] = (64 << 16);          // tensor_dim1 hi=0; tile_dim0=64 dwords (127:112)
  g1[4] = 32;                  // tile_dim1=32 rows; tile_dim2=0
  g1[5] = row_stride_dwords;   // tensor_dim0_stride[31:0] (bits 191:160)
  g1[6] = 0;                   // stride hi, tensor_dim1_stride lo
  g1[7] = 0;

  const i32x4 gz4 = {0, 0, 0, 0};              // groups 2/3 unused (2-D tensor)
  const i32x8 gz8 = {0, 0, 0, 0, 0, 0, 0, 0};  // extra group (clang-23 6-arg form)
  __builtin_amdgcn_tensor_load_to_lds(g0, g1, gz4, gz4, gz8, 0);
}

// =====================================================================
// Flash attention with causal mask + GQA (head h -> latent head h/4).
// Grid: (S/128 query tiles, B*NH). Block: 256 threads, wave owns 16 rows.
// Q: [B*S, NH*HD] bf16.  K/V: [(b*S+s)*NLH+g][HD] bf16.  O: [B*S, NH*HD].
// K/V tiles staged into LDS by the Tensor Data Mover (TENSORcnt).
// =====================================================================
__global__ __launch_bounds__(256)
void mla_flash_attn(const __bf16* __restrict__ Q,
                    const __bf16* __restrict__ Kv,
                    const __bf16* __restrict__ Vv,
                    __bf16* __restrict__ O)
{
  const int qt = blockIdx.x;
  const int bh = blockIdx.y;
  const int b  = bh / NHc;
  const int h  = bh % NHc;
  const int g  = h / (NHc / NLHc);

  const int tid  = threadIdx.x;
  const int wave = tid >> 5;
  const int lane = tid & 31;
  const int half = lane >> 4;
  const int l16  = lane & 15;

  __shared__ __bf16 Ks[32][HDc + 2];
  __shared__ __bf16 Vs[32][HDc + 2];
  __shared__ __bf16 Ps[8][16][34];

  const unsigned lds_k = (unsigned)(size_t)(&Ks[0][0]);  // LDS aperture: low 32b = offset
  const unsigned lds_v = (unsigned)(size_t)(&Vs[0][0]);

  // Resident Q fragments: 16 rows x 128 hd as 4 k-chunks of 32
  v16bf qf[4];
  const int qrow = qt * 128 + (wave << 4) + l16;
  #pragma unroll
  for (int c = 0; c < 4; ++c) {
    #pragma unroll
    for (int j = 0; j < 16; ++j) {
      const int kk = c * 32 + ((j >> 3) << 4) + (half << 3) + (j & 7);
      qf[c][j] = Q[(size_t)(b * Sc + qrow) * (NHc * HDc) + h * HDc + kk];
    }
  }

  v8f acco[8] = {};
  float mrow[8], lrow[8];
  #pragma unroll
  for (int r = 0; r < 8; ++r) { mrow[r] = -3.0e38f; lrow[r] = 0.f; }

  const float scale = 0.088388347648318447f;   // 1/sqrt(128)
  const int ktiles = (qt + 1) * (128 / 32);    // causal: only keys <= q tile end

  for (int kt = 0; kt < ktiles; ++kt) {
    __syncthreads();   // previous tile fully consumed by all waves
    if (wave == 0) {
      const size_t kvrow = ((size_t)(b * Sc + kt * 32) * NLHc + g) * HDc;
      // row stride between consecutive keys: NLH*HD bf16 = 1024 B = 256 dwords
      tdm_load_tile_32x128bf16(Kv + kvrow, lds_k, (NLHc * HDc * 2) / 4);
      tdm_load_tile_32x128bf16(Vv + kvrow, lds_v, (NLHc * HDc * 2) / 4);
      __builtin_amdgcn_s_wait_tensorcnt(0);
    }
    __syncthreads();   // tiles visible to all waves

    // S = Q @ K^T : [16 x 32], contraction over HD in 4 chunks of 32
    v8f s[2] = {};
    #pragma unroll
    for (int c = 0; c < 4; ++c) {
      #pragma unroll
      for (int t = 0; t < 2; ++t) {
        v16bf kf;   // B[k][n] = K[key = t*16+n][hd = c*32+k]
        #pragma unroll
        for (int j = 0; j < 16; ++j)
          kf[j] = Ks[(t << 4) + l16][c * 32 + ((j >> 3) << 4) + (half << 3) + (j & 7)];
        s[t] = __builtin_amdgcn_wmma_f32_16x16x32_bf16(
            false, qf[c], false, kf, (short)0, s[t], false, false);
      }
    }

    // Online softmax per row (row m = r + 8*half spans one 16-lane half)
    #pragma unroll
    for (int r = 0; r < 8; ++r) {
      const int qg = qt * 128 + (wave << 4) + r + (half << 3);
      float sv0 = s[0][r] * scale + ((kt * 32 + l16)      <= qg ? 0.f : -1.0e9f);
      float sv1 = s[1][r] * scale + ((kt * 32 + 16 + l16) <= qg ? 0.f : -1.0e9f);
      float mx = fmaxf(sv0, sv1);
      mx = fmaxf(mx, __shfl_xor(mx, 1, 32));
      mx = fmaxf(mx, __shfl_xor(mx, 2, 32));
      mx = fmaxf(mx, __shfl_xor(mx, 4, 32));
      mx = fmaxf(mx, __shfl_xor(mx, 8, 32));
      const float mnew = fmaxf(mrow[r], mx);
      const float corr = __expf(mrow[r] - mnew);
      const float p0 = __expf(sv0 - mnew);
      const float p1 = __expf(sv1 - mnew);
      float rs = p0 + p1;
      rs += __shfl_xor(rs, 1, 32);
      rs += __shfl_xor(rs, 2, 32);
      rs += __shfl_xor(rs, 4, 32);
      rs += __shfl_xor(rs, 8, 32);
      lrow[r] = lrow[r] * corr + rs;
      mrow[r] = mnew;
      #pragma unroll
      for (int t = 0; t < 8; ++t) acco[t][r] = acco[t][r] * corr;
      Ps[wave][(half << 3) + r][l16]      = (__bf16)p0;
      Ps[wave][(half << 3) + r][16 + l16] = (__bf16)p1;
    }

    // Re-lay P into A-fragment form (wave-private LDS; same-wave DS is ordered)
    v16bf pf;
    #pragma unroll
    for (int j = 0; j < 16; ++j) {
      const int kk = ((j >> 3) << 4) + (half << 3) + (j & 7);   // key index
      pf[j] = Ps[wave][l16][kk];
    }
    // O += P @ V : contraction over 32 keys, 8 output chunks of 16 hd
    #pragma unroll
    for (int t = 0; t < 8; ++t) {
      v16bf vf;   // B[k][n] = V[key = k][hd = t*16+n]
      #pragma unroll
      for (int j = 0; j < 16; ++j)
        vf[j] = Vs[((j >> 3) << 4) + (half << 3) + (j & 7)][(t << 4) + l16];
      acco[t] = __builtin_amdgcn_wmma_f32_16x16x32_bf16(
          false, pf, false, vf, (short)0, acco[t], false, false);
    }
  }

  // Epilogue: normalize by row sum, store bf16 for the output projection
  #pragma unroll
  for (int t = 0; t < 8; ++t) {
    #pragma unroll
    for (int r = 0; r < 8; ++r) {
      const int m  = r + (half << 3);
      const int qg = qt * 128 + (wave << 4) + m;
      const float ov = acco[t][r] / lrow[r];
      O[(size_t)(b * Sc + qg) * (NHc * HDc) + h * HDc + (t << 4) + l16] = (__bf16)ov;
    }
  }
}

// =====================================================================
extern "C" void kernel_launch(void* const* d_in, const int* in_sizes, int n_in,
                              void* d_out, int out_size, void* d_ws, size_t ws_size,
                              hipStream_t stream)
{
  (void)in_sizes; (void)n_in; (void)out_size; (void)ws_size;
  const float* hidden = (const float*)d_in[0];
  // d_in[1] position_ids: RoPE is identity here.  d_in[2] mask: exact causal
  // -1e9 mask, reproduced analytically in the attention kernel.
  const float* q_w  = (const float*)d_in[3];
  const float* q_b  = (const float*)d_in[4];
  const float* kl_w = (const float*)d_in[5];
  const float* kl_b = (const float*)d_in[6];
  const float* vl_w = (const float*)d_in[7];
  const float* vl_b = (const float*)d_in[8];
  const float* kr_w = (const float*)d_in[9];
  const float* kr_b = (const float*)d_in[10];
  const float* vr_w = (const float*)d_in[11];
  const float* vr_b = (const float*)d_in[12];
  const float* o_w  = (const float*)d_in[13];
  const float* o_b  = (const float*)d_in[14];
  float* out = (float*)d_out;

  char* ws = (char*)d_ws;
  const size_t nBS    = (size_t)Bc * Sc;                       // 4096
  const size_t sz_q   = nBS * (size_t)Hc * sizeof(__bf16);     // 16 MB
  const size_t sz_lat = nBS * (size_t)(NLHc * LDc) * sizeof(__bf16);
  const size_t sz_kv  = nBS * (size_t)(NLHc * HDc) * sizeof(__bf16);
  __bf16* ws_q    = (__bf16*)(ws);
  __bf16* ws_klat = (__bf16*)(ws + sz_q);
  __bf16* ws_vlat = (__bf16*)(ws + sz_q + sz_lat);
  __bf16* ws_k    = (__bf16*)(ws + sz_q + 2 * sz_lat);
  __bf16* ws_v    = (__bf16*)(ws + sz_q + 2 * sz_lat + sz_kv);
  __bf16* ws_attn = (__bf16*)(ws + sz_q + 2 * sz_lat + 2 * sz_kv);

  const dim3 blk(256);
  const int M = (int)nBS;                       // 4096 token rows

  // Projections (M=4096, N=2048, K=2048)
  gemm_bias_wmma<float, __bf16><<<dim3(Hc / 128, M / 128), blk, 0, stream>>>(
      hidden, q_w, q_b, ws_q, M, NHc * HDc, Hc);
  gemm_bias_wmma<float, __bf16><<<dim3((NLHc * LDc) / 128, M / 128), blk, 0, stream>>>(
      hidden, kl_w, kl_b, ws_klat, M, NLHc * LDc, Hc);
  gemm_bias_wmma<float, __bf16><<<dim3((NLHc * LDc) / 128, M / 128), blk, 0, stream>>>(
      hidden, vl_w, vl_b, ws_vlat, M, NLHc * LDc, Hc);

  // Latent restore: view k_lat as [B*S*NLH, 512] (contiguous reshape),
  // M=16384, N=128, K=512 — weight shared across latent heads.
  const int Ml = (int)(nBS * NLHc);
  gemm_bias_wmma<__bf16, __bf16><<<dim3(HDc / 128, Ml / 128), blk, 0, stream>>>(
      ws_klat, kr_w, kr_b, ws_k, Ml, HDc, LDc);
  gemm_bias_wmma<__bf16, __bf16><<<dim3(HDc / 128, Ml / 128), blk, 0, stream>>>(
      ws_vlat, vr_w, vr_b, ws_v, Ml, HDc, LDc);

  // Causal flash attention with GQA (TDM-staged K/V tiles)
  mla_flash_attn<<<dim3(Sc / 128, Bc * NHc), blk, 0, stream>>>(
      ws_q, ws_k, ws_v, ws_attn);

  // Output projection -> f32 result
  gemm_bias_wmma<__bf16, float><<<dim3(Hc / 128, M / 128), blk, 0, stream>>>(
      ws_attn, o_w, o_b, out, M, Hc, Hc);
}
